// LSTMLayer_12292196401868
// MI455X (gfx1250) — matrix operand order repaired and checked by reading, not compiled, not run on hardware
//
#include <hip/hip_runtime.h>

// ---------------------------------------------------------------------------
// LSTM layer on gfx1250 (MI455X): persistent bf16-WMMA recurrence kernel.
// B=128, T=256, F=512, H=1024.  z_t = [x_t | h_{t-1}] @ [W;U]^T + b
// One launch, 256 timesteps, grid-wide atomic barrier per step, cell state
// held in registers for the entire sequence.
// ---------------------------------------------------------------------------

#define LSTM_B 128
#define LSTM_T 256
#define LSTM_F 512
#define LSTM_H 1024
#define LSTM_K (LSTM_F + LSTM_H)   // 1536
#define LSTM_G4 (4 * LSTM_H)       // 4096
#define LSTM_NWG 64                // persistent workgroups (4 m-tiles x 16 n-tiles)

typedef __attribute__((ext_vector_type(16))) __bf16 v16bf;
typedef __attribute__((ext_vector_type(8)))  float  v8f;
typedef unsigned int u32x4 __attribute__((ext_vector_type(4)));

union Frag16 {
    u32x4 q[2];
    v16bf v;
};

__device__ __forceinline__ unsigned short f32_to_bf16(float f) {
    unsigned int u = __float_as_uint(f);
    unsigned int r = 0x7FFFu + ((u >> 16) & 1u);   // round-to-nearest-even
    u += r;
    return (unsigned short)(u >> 16);
}

__device__ __forceinline__ float sigmoidf_(float x) {
    return 1.0f / (1.0f + __expf(-x));
}
__device__ __forceinline__ float tanhf_(float x) {
    return 1.0f - 2.0f / (__expf(2.0f * x) + 1.0f);
}

// --------------------------- prep kernels ----------------------------------

// x [B,T,F] f32  ->  xb [T,B,F] bf16
__global__ void lstm_convert_x(const float* __restrict__ x,
                               unsigned short* __restrict__ xb) {
    int i = blockIdx.x * blockDim.x + threadIdx.x;
    const int total = LSTM_B * LSTM_T * LSTM_F;
    if (i >= total) return;
    int f  = i % LSTM_F;
    int bt = i / LSTM_F;
    int b  = bt % LSTM_B;
    int t  = bt / LSTM_B;
    xb[i] = f32_to_bf16(x[((size_t)b * LSTM_T + t) * LSTM_F + f]);
}

// W [F,4H], U [H,4H] f32  ->  Wt [4H, K] bf16  (transposed, K contiguous)
__global__ void lstm_convert_w(const float* __restrict__ W,
                               const float* __restrict__ U,
                               unsigned short* __restrict__ Wt) {
    int i = blockIdx.x * blockDim.x + threadIdx.x;
    const int total = LSTM_G4 * LSTM_K;
    if (i >= total) return;
    int n = i % LSTM_G4;
    int k = i / LSTM_G4;
    float v = (k < LSTM_F) ? W[(size_t)k * LSTM_G4 + n]
                           : U[(size_t)(k - LSTM_F) * LSTM_G4 + n];
    Wt[(size_t)n * LSTM_K + k] = f32_to_bf16(v);
}

// zero h0 (bf16) and the grid-barrier counter
__global__ void lstm_init_state(unsigned short* __restrict__ h0,
                                unsigned int* __restrict__ barrier_ctr) {
    int i = blockIdx.x * blockDim.x + threadIdx.x;
    if (i == 0) *barrier_ctr = 0u;
    if (i >= LSTM_B * LSTM_H) return;
    h0[i] = 0;
}

// --------------------------- persistent recurrence -------------------------
// grid: 64 WGs = (4 batch tiles of 32) x (16 hidden tiles of 64).
// block: 128 threads = 4 waves (wave32). Wave w owns hidden sub-tile w (16
// cols) and BOTH 16-row batch sub-tiles -> 8 accumulators (2 m x 4 gates).
// Cell state c lives in registers for all 256 steps.
__global__ __launch_bounds__(128) void lstm_persistent(
    const unsigned short* __restrict__ xb,     // [T, B, F] bf16
    const unsigned short* __restrict__ Wt,     // [4H, K] bf16 transposed
    const float* __restrict__ bias,            // [4H] f32, gate order i,f,g,o
    unsigned short* __restrict__ hbuf0,        // [B, H] bf16 (holds h_{-1}=0)
    unsigned short* __restrict__ hbuf1,        // [B, H] bf16
    float* __restrict__ out,                   // [T, B, H] f32
    unsigned int* __restrict__ barrier_ctr)
{
    const int tid  = threadIdx.x;
    const int wave = tid >> 5;    // 0..3: hidden sub-tile within WG
    const int lane = tid & 31;
    const int hl   = lane >> 4;   // half-wave selects K sub-chunk / row group
    const int idx  = lane & 15;

    const int mt = blockIdx.x >> 4;       // 0..3  batch tile (32 rows)
    const int nt = blockIdx.x & 15;       // 0..15 hidden tile (64 cols)
    const int m0 = mt * 32;
    const int n  = nt * 64 + wave * 16 + idx;   // hidden column this lane owns

    // rows this lane loads for the two A fragments
    const int mA0 = m0 + idx;
    const int mA1 = m0 + 16 + idx;

    const unsigned short* brow[4];
#pragma unroll
    for (int g = 0; g < 4; ++g)
        brow[g] = Wt + (size_t)(g * LSTM_H + n) * LSTM_K;

    const float bi = bias[0 * LSTM_H + n];
    const float bf = bias[1 * LSTM_H + n];
    const float bg = bias[2 * LSTM_H + n];
    const float bo = bias[3 * LSTM_H + n];

    // register-resident cell state: 2 m-subtiles x 8 rows per lane
    float creg0[8] = {}, creg1[8] = {};

    for (int t = 0; t < LSTM_T; ++t) {
        const unsigned short* x_t  = xb + (size_t)t * LSTM_B * LSTM_F;
        const unsigned short* h_in = (t & 1) ? hbuf1 : hbuf0;
        unsigned short*      h_out = (t & 1) ? hbuf0 : hbuf1;

        const unsigned short* ax0 = x_t + (size_t)mA0 * LSTM_F;
        const unsigned short* ax1 = x_t + (size_t)mA1 * LSTM_F;
        const unsigned short* ah0 = h_in + (size_t)mA0 * LSTM_H;
        const unsigned short* ah1 = h_in + (size_t)mA1 * LSTM_H;

        v8f acc0[4] = {v8f{}, v8f{}, v8f{}, v8f{}};
        v8f acc1[4] = {v8f{}, v8f{}, v8f{}, v8f{}};

        // ---- K over the x_t @ W part: k = 0 .. F-1 ----
#pragma unroll 4
        for (int k0 = 0; k0 < LSTM_F; k0 += 32) {
            Frag16 a0, a1, b[4];
            a0.q[0] = *(const u32x4*)(ax0 + k0 + hl * 8);
            a0.q[1] = *(const u32x4*)(ax0 + k0 + 16 + hl * 8);
            a1.q[0] = *(const u32x4*)(ax1 + k0 + hl * 8);
            a1.q[1] = *(const u32x4*)(ax1 + k0 + 16 + hl * 8);
#pragma unroll
            for (int g = 0; g < 4; ++g) {
                b[g].q[0] = *(const u32x4*)(brow[g] + k0 + hl * 8);
                b[g].q[1] = *(const u32x4*)(brow[g] + k0 + 16 + hl * 8);
            }
#pragma unroll
            for (int g = 0; g < 4; ++g) {
                acc0[g] = __builtin_amdgcn_wmma_f32_16x16x32_bf16(
                    false, a0.v, false, b[g].v, (short)0, acc0[g], false, false);
                acc1[g] = __builtin_amdgcn_wmma_f32_16x16x32_bf16(
                    false, a1.v, false, b[g].v, (short)0, acc1[g], false, false);
            }
        }

        // ---- K over the h_{t-1} @ U part: k = F .. F+H-1 ----
#pragma unroll 4
        for (int k0 = 0; k0 < LSTM_H; k0 += 32) {
            Frag16 a0, a1, b[4];
            a0.q[0] = *(const u32x4*)(ah0 + k0 + hl * 8);
            a0.q[1] = *(const u32x4*)(ah0 + k0 + 16 + hl * 8);
            a1.q[0] = *(const u32x4*)(ah1 + k0 + hl * 8);
            a1.q[1] = *(const u32x4*)(ah1 + k0 + 16 + hl * 8);
            const int kw = LSTM_F + k0;
#pragma unroll
            for (int g = 0; g < 4; ++g) {
                b[g].q[0] = *(const u32x4*)(brow[g] + kw + hl * 8);
                b[g].q[1] = *(const u32x4*)(brow[g] + kw + 16 + hl * 8);
            }
#pragma unroll
            for (int g = 0; g < 4; ++g) {
                acc0[g] = __builtin_amdgcn_wmma_f32_16x16x32_bf16(
                    false, a0.v, false, b[g].v, (short)0, acc0[g], false, false);
                acc1[g] = __builtin_amdgcn_wmma_f32_16x16x32_bf16(
                    false, a1.v, false, b[g].v, (short)0, acc1[g], false, false);
            }
        }

        // ---- epilogue: bias, gates, register-resident cell update ----
        // C/D layout: lane owns column n; VGPR r holds row base + 8*hl + r.
        float* outT = out + (size_t)t * LSTM_B * LSTM_H;
#pragma unroll
        for (int r = 0; r < 8; ++r) {
            {   // m sub-tile 0
                const int mm = m0 + hl * 8 + r;
                const size_t ix = (size_t)mm * LSTM_H + n;
                const float ig = sigmoidf_(acc0[0][r] + bi);
                const float fg = sigmoidf_(acc0[1][r] + bf);
                const float gg = tanhf_(acc0[2][r] + bg);
                const float og = sigmoidf_(acc0[3][r] + bo);
                const float cnew = fg * creg0[r] + ig * gg;
                const float hnew = og * tanhf_(cnew);
                creg0[r] = cnew;
                outT[ix] = hnew;
                h_out[ix] = f32_to_bf16(hnew);
            }
            {   // m sub-tile 1
                const int mm = m0 + 16 + hl * 8 + r;
                const size_t ix = (size_t)mm * LSTM_H + n;
                const float ig = sigmoidf_(acc1[0][r] + bi);
                const float fg = sigmoidf_(acc1[1][r] + bf);
                const float gg = tanhf_(acc1[2][r] + bg);
                const float og = sigmoidf_(acc1[3][r] + bo);
                const float cnew = fg * creg1[r] + ig * gg;
                const float hnew = og * tanhf_(cnew);
                creg1[r] = cnew;
                outT[ix] = hnew;
                h_out[ix] = f32_to_bf16(hnew);
            }
        }

        // ---- grid-wide barrier: make h_out visible, wait for all 64 WGs ----
        __threadfence();            // release: flush h_out to device scope
        __syncthreads();
        if (tid == 0) {
            atomicAdd(barrier_ctr, 1u);
            const unsigned int target = (unsigned int)LSTM_NWG * (unsigned int)(t + 1);
            while (__hip_atomic_load(barrier_ctr, __ATOMIC_ACQUIRE,
                                     __HIP_MEMORY_SCOPE_AGENT) < target) {
                __builtin_amdgcn_s_sleep(1);
            }
        }
        __syncthreads();
        __threadfence();            // acquire: invalidate caches before next read
    }
}

// --------------------------- host launcher ---------------------------------

extern "C" void kernel_launch(void* const* d_in, const int* in_sizes, int n_in,
                              void* d_out, int out_size, void* d_ws, size_t ws_size,
                              hipStream_t stream) {
    const float* x = (const float*)d_in[0];   // [B,T,F]
    const float* W = (const float*)d_in[1];   // [F,4H]
    const float* U = (const float*)d_in[2];   // [H,4H]
    const float* b = (const float*)d_in[3];   // [4H]
    float* out = (float*)d_out;               // [T,B,H]

    // workspace layout (bytes):
    //   xb : T*B*F bf16      = 33,554,432
    //   Wt : 4H*K bf16       = 12,582,912
    //   h0 : B*H bf16        =    262,144
    //   h1 : B*H bf16        =    262,144
    //   ctr: barrier counter (4KB slot)          (total ~46.7 MB)
    char* ws = (char*)d_ws;
    unsigned short* xb = (unsigned short*)(ws);
    unsigned short* Wt = (unsigned short*)(ws + 33554432u);
    unsigned short* h0 = (unsigned short*)(ws + 33554432u + 12582912u);
    unsigned short* h1 = (unsigned short*)(ws + 33554432u + 12582912u + 262144u);
    unsigned int*  ctr = (unsigned int*)(ws + 33554432u + 12582912u + 2u * 262144u);

    {
        int nx = LSTM_B * LSTM_T * LSTM_F;
        lstm_convert_x<<<(nx + 255) / 256, 256, 0, stream>>>(x, xb);
    }
    {
        int nw = LSTM_G4 * LSTM_K;
        lstm_convert_w<<<(nw + 255) / 256, 256, 0, stream>>>(W, U, Wt);
    }
    {
        int ns = LSTM_B * LSTM_H;
        lstm_init_state<<<(ns + 255) / 256, 256, 0, stream>>>(h0, ctr);
    }

    lstm_persistent<<<LSTM_NWG, 128, 0, stream>>>(xb, Wt, b, h0, h1, out, ctr);
}
